// _MHLinearAttn_4758823764279
// MI455X (gfx1250) — matrix operand order repaired and checked by reading, not compiled, not run on hardware
//
#include <hip/hip_runtime.h>
#include <hip/hip_bf16.h>

// ---------------- problem constants ----------------
#define B_    4
#define T_    2048
#define H_    16
#define DH_   64
#define D_    1024
#define NC_   32          // chunks per (b,h):  T / 64
#define CHK_  64          // chunk length
#define EPS_  1e-6f

typedef __bf16 v8bf  __attribute__((ext_vector_type(8)));
typedef __bf16 v16bf __attribute__((ext_vector_type(16)));
typedef float  v8f   __attribute__((ext_vector_type(8)));

union f16x16u { v16bf v; v8bf h[2]; };

__device__ __forceinline__ v8f wmma_bf16(v16bf a, v16bf b, v8f c) {
  // D = A(16x32 bf16) * B(32x16 bf16) + C(16x16 f32)
  return __builtin_amdgcn_wmma_f32_16x16x32_bf16(false, a, false, b, (short)0, c,
                                                 false, false);
}

// ---------------- elementwise fp32 -> bf16 ----------------
__global__ __launch_bounds__(256) void k_f32_to_bf16(const float* __restrict__ in,
                                                     __bf16* __restrict__ out, int n) {
  int i = blockIdx.x * 256 + threadIdx.x;
  if (i < n) out[i] = (__bf16)in[i];
}

// ---------------- generic NT GEMM: C[M,N] = A[M,K] @ Bt[N,K]^T ----------------
// block = 256 threads = 8 waves; block tile 128(M) x 64(N); wave tile 16 x 64.
__global__ __launch_bounds__(256)
void k_gemm_nt(const __bf16* __restrict__ A, const __bf16* __restrict__ Bt,
               float* __restrict__ C, int M, int N, int K) {
  (void)M;
  int tid  = threadIdx.x;
  int wave = tid >> 5;
  int lane = tid & 31;
  int l    = lane & 15;
  int h8   = (lane >> 4) << 3;   // A-frag K sub-offset {0,8}; also D row offset
  int kB   = (lane >> 4) << 4;   // unused; B-frag loads use h8 like A

  int m0 = blockIdx.y * 128 + wave * 16;
  int n0 = blockIdx.x * 64;

  v8f acc[4] = {};
  const __bf16* arow = A + (size_t)(m0 + l) * K;
  for (int k0 = 0; k0 < K; k0 += 32) {
    f16x16u af;
    af.h[0] = *(const v8bf*)(arow + k0 + h8);
    af.h[1] = *(const v8bf*)(arow + k0 + 16 + h8);
#pragma unroll
    for (int j = 0; j < 4; ++j) {
      const __bf16* brow = Bt + (size_t)(n0 + 16 * j + l) * K;
      f16x16u bf;
      bf.h[0] = *(const v8bf*)(brow + k0 + h8);
      bf.h[1] = *(const v8bf*)(brow + k0 + 16 + h8);
      acc[j] = wmma_bf16(af.v, bf.v, acc[j]);
    }
  }
#pragma unroll
  for (int j = 0; j < 4; ++j)
#pragma unroll
    for (int i = 0; i < 8; ++i)
      C[(size_t)(m0 + i + h8) * N + n0 + 16 * j + l] = acc[j][i];
}

// ---------------- RoPE + phi(elu+1), relayout [B,T,H*64] -> [B,H,T,64] bf16 ---
__global__ __launch_bounds__(256)
void k_rope_phi(const float* __restrict__ R, __bf16* __restrict__ P) {
  int idx = blockIdx.x * 256 + threadIdx.x;      // (b,t,h,d<32)
  int d = idx & 31;
  int hh = (idx >> 5) & 15;
  int t = (idx >> 9) & (T_ - 1);
  int b = idx >> 20;
  const float* row = R + (size_t)(b * T_ + t) * D_ + hh * 64;
  float x1 = row[d], x2 = row[d + 32];
  // inv_freq = 10000^(-d/32) = exp(-d * ln(1e4)/32)
  float inv = __expf(-(float)d * (9.210340371976184f / 32.0f));
  float ang = (float)t * inv;
  float s, c;
  __sincosf(ang, &s, &c);
  float r1 = x1 * c - x2 * s;
  float r2 = x1 * s + x2 * c;
  r1 = r1 > 0.f ? r1 + 1.f : __expf(r1);          // elu(x)+1
  r2 = r2 > 0.f ? r2 + 1.f : __expf(r2);
  __bf16* orow = P + (size_t)((b * H_ + hh) * T_ + t) * 64;
  orow[d] = (__bf16)r1;
  orow[d + 32] = (__bf16)r2;
}

__global__ __launch_bounds__(256)
void k_relayout_v(const float* __restrict__ R, __bf16* __restrict__ P) {
  int idx = blockIdx.x * 256 + threadIdx.x;      // (b,t,h,d<64)
  int d = idx & 63;
  int hh = (idx >> 6) & 15;
  int t = (idx >> 10) & (T_ - 1);
  int b = idx >> 21;
  P[(size_t)((b * H_ + hh) * T_ + t) * 64 + d] =
      (__bf16)R[(size_t)(b * T_ + t) * D_ + hh * 64 + d];
}

// ---------------- per-chunk state: ST[e][d] = sum_t V[t][e]*K[t][d] ----------
// (ST is the transposed KV increment so later B-fragments are row-contiguous)
__global__ __launch_bounds__(128)
void k_chunk_state(const __bf16* __restrict__ Kb, const __bf16* __restrict__ Vb,
                   float* __restrict__ ST, float* __restrict__ Ksum) {
  __shared__ __align__(16) __bf16 Kt[64][72];
  __shared__ __align__(16) __bf16 Vt[64][72];
  int blk = blockIdx.x;                    // bh*NC + c
  int c = blk & (NC_ - 1);
  int bh = blk >> 5;
  const __bf16* kc = Kb + (size_t)(bh * T_ + c * CHK_) * 64;
  const __bf16* vc = Vb + (size_t)(bh * T_ + c * CHK_) * 64;
  int tid = threadIdx.x;
  for (int i = tid; i < 4096; i += 128) {  // transpose-stage into LDS
    int t = i >> 6, d = i & 63;
    Kt[d][t] = kc[i];
    Vt[d][t] = vc[i];
  }
  __syncthreads();
  if (tid < 64) {                          // column sums of K-phi
    float s = 0.f;
    for (int t = 0; t < 64; ++t) s += (float)Kt[tid][t];
    Ksum[(size_t)blk * 64 + tid] = s;
  }
  int wave = tid >> 5, lane = tid & 31, l = lane & 15;
  int h8 = (lane >> 4) << 3, kB = (lane >> 4) << 4;
  int e0 = wave * 16;
  v8f acc[4] = {};
  for (int k0 = 0; k0 < 64; k0 += 32) {
    f16x16u af;
    af.h[0] = *(const v8bf*)(&Vt[e0 + l][k0 + h8]);
    af.h[1] = *(const v8bf*)(&Vt[e0 + l][k0 + 16 + h8]);
#pragma unroll
    for (int j = 0; j < 4; ++j) {
      f16x16u bf;
      bf.h[0] = *(const v8bf*)(&Kt[16 * j + l][k0 + h8]);
      bf.h[1] = *(const v8bf*)(&Kt[16 * j + l][k0 + 16 + h8]);
      acc[j] = wmma_bf16(af.v, bf.v, acc[j]);
    }
  }
  float* out = ST + (size_t)blk * 4096;
#pragma unroll
  for (int j = 0; j < 4; ++j)
#pragma unroll
    for (int i = 0; i < 8; ++i)
      out[(e0 + i + h8) * 64 + 16 * j + l] = acc[j][i];
}

// ---------------- exclusive prefix scan over the 32 chunk states -------------
__global__ __launch_bounds__(256)
void k_scan(const float* __restrict__ ST, const float* __restrict__ Ksum,
            __bf16* __restrict__ KVT, float* __restrict__ KsumP) {
  int bh = blockIdx.x;
  int tid = threadIdx.x;
  float acc[16];
#pragma unroll
  for (int j = 0; j < 16; ++j) acc[j] = 0.f;
  float sacc = 0.f;
  for (int c = 0; c < NC_; ++c) {
    size_t base = ((size_t)bh * NC_ + c) * 4096;
#pragma unroll
    for (int j = 0; j < 16; ++j) {
      int idx = tid + 256 * j;
      KVT[base + idx] = (__bf16)acc[j];
      acc[j] += ST[base + idx];
    }
    if (tid < 64) {
      size_t kb = ((size_t)bh * NC_ + c) * 64 + tid;
      KsumP[kb] = sacc;
      sacc += Ksum[kb];
    }
  }
}

// ---------------- chunk attention: Y = mask(Q K^T) V + Q KV_prev, / den ------
__global__ __launch_bounds__(128)
void k_attn_chunk(const __bf16* __restrict__ Qb, const __bf16* __restrict__ Kb,
                  const __bf16* __restrict__ Vb, const __bf16* __restrict__ KVT,
                  const float* __restrict__ KsumP, float* __restrict__ Y) {
  __shared__ __align__(16) __bf16 Vt[64][72];
  __shared__ __align__(16) __bf16 Pm[64][72];
  __shared__ float den[64];
  __shared__ float ksp[64];
  int blk = blockIdx.x;
  int c = blk & (NC_ - 1);
  int bh = blk >> 5;
  int b = bh >> 4, hh = bh & 15;
  size_t cbase = (size_t)(bh * T_ + c * CHK_) * 64;
  const __bf16* qc = Qb + cbase;
  const __bf16* kc = Kb + cbase;
  const __bf16* vc = Vb + cbase;
  const __bf16* kv = KVT + (size_t)blk * 4096;
  int tid = threadIdx.x;
  for (int i = tid; i < 4096; i += 128) {  // stage V transposed
    int t = i >> 6, e = i & 63;
    Vt[e][t] = vc[i];
  }
  if (tid < 64) ksp[tid] = KsumP[(size_t)blk * 64 + tid];
  __syncthreads();

  int wave = tid >> 5, lane = tid & 31, l = lane & 15;
  int h8 = (lane >> 4) << 3, kB = (lane >> 4) << 4;
  int t0 = wave * 16;
  const __bf16* arow = qc + (size_t)(t0 + l) * 64;

  // P = Qphi @ Kphi^T   (K dim = d)
  v8f p[4] = {};
  for (int k0 = 0; k0 < 64; k0 += 32) {
    f16x16u af;
    af.h[0] = *(const v8bf*)(arow + k0 + h8);
    af.h[1] = *(const v8bf*)(arow + k0 + 16 + h8);
#pragma unroll
    for (int j = 0; j < 4; ++j) {
      const __bf16* brow = kc + (size_t)(16 * j + l) * 64;  // col t' of K^T
      f16x16u bf;
      bf.h[0] = *(const v8bf*)(brow + k0 + h8);
      bf.h[1] = *(const v8bf*)(brow + k0 + 16 + h8);
      p[j] = wmma_bf16(af.v, bf.v, p[j]);
    }
  }
  // causal mask within chunk, stash bf16 copy for the second WMMA pass
#pragma unroll
  for (int j = 0; j < 4; ++j)
#pragma unroll
    for (int i = 0; i < 8; ++i) {
      int t = t0 + i + h8, tp = 16 * j + l;
      Pm[t][tp] = (__bf16)((tp <= t) ? p[j][i] : 0.f);
    }
  __syncthreads();
  if (tid < 64) {                          // den[t] = rowsum(Pm) + q . Ksum_prev
    float s = 0.f;
    for (int tp = 0; tp < 64; ++tp) s += (float)Pm[tid][tp];
    const __bf16* qrow = qc + (size_t)tid * 64;
    for (int d = 0; d < 64; ++d) s += (float)qrow[d] * ksp[d];
    den[tid] = s;
  }
  // Y = Pm @ V  (K = t')   +   Qphi @ KV_prev  (K = d)
  v8f acc[4] = {};
  for (int k0 = 0; k0 < 64; k0 += 32) {
    f16x16u af;
    af.h[0] = *(const v8bf*)(&Pm[t0 + l][k0 + h8]);
    af.h[1] = *(const v8bf*)(&Pm[t0 + l][k0 + 16 + h8]);
#pragma unroll
    for (int j = 0; j < 4; ++j) {
      f16x16u bf;
      bf.h[0] = *(const v8bf*)(&Vt[16 * j + l][k0 + h8]);
      bf.h[1] = *(const v8bf*)(&Vt[16 * j + l][k0 + 16 + h8]);
      acc[j] = wmma_bf16(af.v, bf.v, acc[j]);
    }
  }
  for (int k0 = 0; k0 < 64; k0 += 32) {
    f16x16u af;
    af.h[0] = *(const v8bf*)(arow + k0 + h8);
    af.h[1] = *(const v8bf*)(arow + k0 + 16 + h8);
#pragma unroll
    for (int j = 0; j < 4; ++j) {
      const __bf16* brow = kv + (size_t)(16 * j + l) * 64;  // KV_prev^T row e
      f16x16u bf;
      bf.h[0] = *(const v8bf*)(brow + k0 + h8);
      bf.h[1] = *(const v8bf*)(brow + k0 + 16 + h8);
      acc[j] = wmma_bf16(af.v, bf.v, acc[j]);
    }
  }
  __syncthreads();
#pragma unroll
  for (int j = 0; j < 4; ++j)
#pragma unroll
    for (int i = 0; i < 8; ++i) {
      int t = t0 + i + h8, e = 16 * j + l;
      float dv = den[t];
      dv = dv > EPS_ ? dv : EPS_;
      Y[(size_t)(b * T_ + c * CHK_ + t) * D_ + hh * 64 + e] = acc[j][i] / dv;
    }
}

// ---------------- host side ----------------
extern "C" void kernel_launch(void* const* d_in, const int* in_sizes, int n_in,
                              void* d_out, int out_size, void* d_ws, size_t ws_size,
                              hipStream_t stream) {
  (void)in_sizes; (void)n_in; (void)out_size; (void)ws_size;
  const float* x  = (const float*)d_in[0];
  const float* Wq = (const float*)d_in[1];
  const float* Wk = (const float*)d_in[2];
  const float* Wv = (const float*)d_in[3];
  const float* Wo = (const float*)d_in[4];

  const size_t MN = (size_t)B_ * T_ * D_;       // 8,388,608 elements
  const size_t DD = (size_t)D_ * D_;            // 1,048,576
  char* ws = (char*)d_ws;
  size_t off = 0;
  auto carve = [&](size_t bytes) {
    char* p = ws + off;
    off = (off + bytes + 255) & ~(size_t)255;
    return p;
  };
  __bf16* xb    = (__bf16*)carve(MN * 2);
  __bf16* Wqb   = (__bf16*)carve(DD * 2);
  __bf16* Wkb   = (__bf16*)carve(DD * 2);
  __bf16* Wvb   = (__bf16*)carve(DD * 2);
  __bf16* Wob   = (__bf16*)carve(DD * 2);
  float*  tmp32 = (float*)carve(MN * 4);
  __bf16* Qb    = (__bf16*)carve(MN * 2);
  __bf16* Kb    = (__bf16*)carve(MN * 2);
  __bf16* Vb    = (__bf16*)carve(MN * 2);
  float*  ST    = (float*)carve((size_t)B_ * H_ * NC_ * 4096 * 4);
  float*  Ksum  = (float*)carve((size_t)B_ * H_ * NC_ * 64 * 4);
  __bf16* KVT   = (__bf16*)carve((size_t)B_ * H_ * NC_ * 4096 * 2);
  float*  KsumP = (float*)carve((size_t)B_ * H_ * NC_ * 64 * 4);
  __bf16* Yb    = (__bf16*)carve(MN * 2);

  k_f32_to_bf16<<<(int)((MN + 255) / 256), 256, 0, stream>>>(x, xb, (int)MN);
  k_f32_to_bf16<<<(int)((DD + 255) / 256), 256, 0, stream>>>(Wq, Wqb, (int)DD);
  k_f32_to_bf16<<<(int)((DD + 255) / 256), 256, 0, stream>>>(Wk, Wkb, (int)DD);
  k_f32_to_bf16<<<(int)((DD + 255) / 256), 256, 0, stream>>>(Wv, Wvb, (int)DD);
  k_f32_to_bf16<<<(int)((DD + 255) / 256), 256, 0, stream>>>(Wo, Wob, (int)DD);

  dim3 gg(D_ / 64, (B_ * T_) / 128);            // (16, 64)
  const int nrope = B_ * T_ * H_ * 32;          // 4,194,304
  const int nv    = B_ * T_ * H_ * 64;          // 8,388,608

  k_gemm_nt<<<gg, 256, 0, stream>>>(xb, Wqb, tmp32, B_ * T_, D_, D_);
  k_rope_phi<<<nrope / 256, 256, 0, stream>>>(tmp32, Qb);
  k_gemm_nt<<<gg, 256, 0, stream>>>(xb, Wkb, tmp32, B_ * T_, D_, D_);
  k_rope_phi<<<nrope / 256, 256, 0, stream>>>(tmp32, Kb);
  k_gemm_nt<<<gg, 256, 0, stream>>>(xb, Wvb, tmp32, B_ * T_, D_, D_);
  k_relayout_v<<<nv / 256, 256, 0, stream>>>(tmp32, Vb);

  k_chunk_state<<<B_ * H_ * NC_, 128, 0, stream>>>(Kb, Vb, ST, Ksum);
  k_scan<<<B_ * H_, 256, 0, stream>>>(ST, Ksum, KVT, KsumP);
  k_attn_chunk<<<B_ * H_ * NC_, 128, 0, stream>>>(Qb, Kb, Vb, KVT, KsumP, tmp32);

  k_f32_to_bf16<<<(int)((MN + 255) / 256), 256, 0, stream>>>(tmp32, Yb, (int)MN);
  k_gemm_nt<<<gg, 256, 0, stream>>>(Yb, Wob, (float*)d_out, B_ * T_, D_, D_);
}
